// BasicPointNet2_43911745634596
// MI455X (gfx1250) — matrix operand order repaired and checked by reading, not compiled, not run on hardware
//
#include <hip/hip_runtime.h>
#include <hip/hip_bf16.h>
#include <math.h>

// ---------------------------------------------------------------------------
// PointNet++ (single-scale-grouping) forward for MI455X / gfx1250.
// GEMM layers run on v_wmma_f32_16x16x32_f16 (wave32 WMMA), f32 accumulate.
// Weight tiles staged into LDS via GLOBAL_LOAD_ASYNC_TO_LDS_B128 (ASYNCcnt).
// ---------------------------------------------------------------------------

#define BN_SCALE 0.99999500003750f   // 1/sqrt(1+1e-5), eval-mode BatchNorm1d

typedef _Float16 v16h __attribute__((ext_vector_type(16)));
typedef _Float16 h8   __attribute__((ext_vector_type(8)));
typedef float    v8f  __attribute__((ext_vector_type(8)));
typedef int      v4i  __attribute__((ext_vector_type(4)));

static constexpr int BATCH = 16;
static constexpr int NPTS  = 2048;
static constexpr int M1    = 1024;
static constexpr int M2    = 256;

// Async LDS copy path (gfx1250): guard so host pass / older clang still build.
#if defined(__has_builtin)
#  if __has_builtin(__builtin_amdgcn_global_load_async_to_lds_b128) && \
      __has_builtin(__builtin_amdgcn_s_wait_asynccnt)
#    define USE_ASYNC_COPY 1
#  endif
#endif
#ifndef USE_ASYNC_COPY
#  define USE_ASYNC_COPY 0
#endif

// ----- WMMA fragment loaders (layouts per CDNA5 ISA) -------------
// A (16x32 f16, MxK): lane<16 holds row M=lane, K {0..7,16..23}; lane>=16 holds
// row M=lane-16, K {8..15,24..31}.  Two b128 LDS loads per fragment.
__device__ __forceinline__ v16h load_a_frag(const _Float16* A, int pitch,
                                            int mtile, int k0, int lane) {
  const _Float16* p = A + (size_t)((mtile << 4) + (lane & 15)) * pitch
                        + k0 + ((lane >> 4) << 3);
  h8 lo = *(const h8*)p;
  h8 hi = *(const h8*)(p + 16);
  v16h r;
#pragma unroll
  for (int i = 0; i < 8; ++i) { r[i] = lo[i]; r[i + 8] = hi[i]; }
  return r;
}

// B (32x16 f16, KxN) staged N-major in LDS: lane n = col, lanes 0-15 cover
// K {0..7,16..23}, lanes 16-31 cover K {8..15,24..31} (matches A).
__device__ __forceinline__ v16h load_b_frag(const _Float16* Wt, int pitch,
                                            int k0, int lane) {
  const _Float16* p = Wt + (size_t)(lane & 15) * pitch + k0 + ((lane >> 4) << 3);
  h8 lo = *(const h8*)p;
  h8 hi = *(const h8*)(p + 16);
  v16h r;
#pragma unroll
  for (int i = 0; i < 8; ++i) { r[i] = lo[i]; r[i + 8] = hi[i]; }
  return r;
}

// ----- fused MLP layer: H = relu(A @ W + b) * BN, or masked row-max ---------
// Alds:  [nwaves*16][Apitch] f16 (zero padded to Kpad)
// WhT:   global f16 [Nout][Kpad], N-major, zero-padded (pre-converted)
// Hlds:  if non-null, f16 output [rows][Hpitch]
// outpart: if Hlds==null, per-wave masked max -> [nwaves][Nout] f32
__device__ __forceinline__ void fused_layer(
    const _Float16* Alds, int Apitch,
    const _Float16* __restrict__ WhT, const float* __restrict__ bg,
    int Kpad, int Nout,
    _Float16* Hlds, int Hpitch,
    float* outpart, int validRows,
    _Float16* wtile, float bn)
{
  const int tid  = threadIdx.x;
  const int lane = tid & 31;
  const int wid  = tid >> 5;       // wave id == M-tile id
  const int NT   = Nout >> 4;
  const int KT   = Kpad >> 5;
  for (int nt = 0; nt < NT; ++nt) {
    __syncthreads();               // all waves done reading previous wtile
    // Stage the contiguous [16][Kpad] f16 slab for this N-tile into LDS.
    const _Float16* slab = WhT + (size_t)(nt << 4) * Kpad;
    const int nch = Kpad << 1;     // number of 16-byte chunks (16*Kpad/8)
#if USE_ASYNC_COPY
    for (int chunk = tid; chunk < nch; chunk += blockDim.x) {
      __builtin_amdgcn_global_load_async_to_lds_b128(
          (__attribute__((address_space(1))) v4i*)(slab + (chunk << 3)),
          (__attribute__((address_space(3))) v4i*)(wtile + (chunk << 3)),
          0, 0);
    }
    __builtin_amdgcn_s_wait_asynccnt(0);
#else
    for (int chunk = tid; chunk < nch; chunk += blockDim.x)
      *(h8*)(wtile + (chunk << 3)) = *(const h8*)(slab + (chunk << 3));
#endif
    __syncthreads();
    v8f acc = {};
    for (int kt = 0; kt < KT; ++kt) {
      v16h a = load_a_frag(Alds, Apitch, wid, kt << 5, lane);
      v16h b = load_b_frag(wtile, Kpad, kt << 5, lane);
      acc = __builtin_amdgcn_wmma_f32_16x16x32_f16(
          false, a, false, b, (short)0, acc, false, false);
    }
    float bias = bg[(nt << 4) + (lane & 15)];
    if (Hlds) {
#pragma unroll
      for (int v = 0; v < 8; ++v) {
        int row = (wid << 4) + ((lane >> 4) << 3) + v;   // C/D layout
        float hv = fmaxf(acc[v] + bias, 0.0f) * bn;
        Hlds[(size_t)row * Hpitch + (nt << 4) + (lane & 15)] = (_Float16)hv;
      }
    } else {
      float mx = -3.0e38f;
#pragma unroll
      for (int v = 0; v < 8; ++v) {
        int row = (wid << 4) + ((lane >> 4) << 3) + v;
        float hv = fmaxf(acc[v] + bias, 0.0f) * bn;
        if (row < validRows) mx = fmaxf(mx, hv);
      }
      mx = fmaxf(mx, __shfl_xor(mx, 16, 32));  // fold lane halves (same N col)
      mx = fmaxf(mx, 0.0f);                    // relu outputs >= 0; safe floor
      if (lane < 16) outpart[wid * Nout + (nt << 4) + lane] = mx;
    }
  }
  __syncthreads();
}

// ----- weight pre-conversion: f32 [Kin][Nout] -> f16 [Nout][Kpad] (padded) --
__global__ void convertw_kernel(const float* __restrict__ W, int Kin, int Nout,
                                int Kpad, _Float16* __restrict__ out)
{
  int i = blockIdx.x * blockDim.x + threadIdx.x;
  int total = Nout * Kpad;
  if (i >= total) return;
  int n = i / Kpad, k = i - n * Kpad;
  out[i] = (_Float16)((k < Kin) ? W[(size_t)k * Nout + n] : 0.0f);
}

// ----- farthest point sampling (one block per cloud) ------------------------
__global__ void __launch_bounds__(256) fps_kernel(
    const float* __restrict__ pos, int n, int m, float* __restrict__ ctr)
{
  const int b = blockIdx.x;
  const float* P = pos + (size_t)b * n * 3;
  float* C = ctr + (size_t)b * m * 3;
  __shared__ float d[NPTS];
  __shared__ float sval[256];
  __shared__ int   sidx[256];
  __shared__ int   bestIdx;
  const int tid = threadIdx.x;
  float x0 = P[0], y0 = P[1], z0 = P[2];
  for (int j = tid; j < n; j += 256) {
    float dx = P[j*3] - x0, dy = P[j*3+1] - y0, dz = P[j*3+2] - z0;
    d[j] = dx*dx + dy*dy + dz*dz;
  }
  if (tid == 0) { C[0] = x0; C[1] = y0; C[2] = z0; }
  __syncthreads();
  for (int i = 1; i < m; ++i) {
    float bv = -1.0f; int bi = 0;
    for (int j = tid; j < n; j += 256) {
      float v = d[j];
      if (v > bv) { bv = v; bi = j; }
    }
    sval[tid] = bv; sidx[tid] = bi;
    __syncthreads();
    for (int s = 128; s > 0; s >>= 1) {
      if (tid < s) {
        float ov = sval[tid + s]; int oi = sidx[tid + s];
        if (ov > sval[tid] || (ov == sval[tid] && oi < sidx[tid])) {
          sval[tid] = ov; sidx[tid] = oi;
        }
      }
      __syncthreads();
    }
    if (tid == 0) bestIdx = sidx[0];
    __syncthreads();
    int bidx = bestIdx;
    float bx = P[bidx*3], by = P[bidx*3+1], bz = P[bidx*3+2];
    if (tid == 0) { C[i*3] = bx; C[i*3+1] = by; C[i*3+2] = bz; }
    for (int j = tid; j < n; j += 256) {
      float dx = P[j*3] - bx, dy = P[j*3+1] - by, dz = P[j*3+2] - bz;
      d[j] = fminf(d[j], dx*dx + dy*dy + dz*dz);
    }
    __syncthreads();
  }
}

// ----- radius neighbor search: one wave32 per query center ------------------
__global__ void __launch_bounds__(128) radius_kernel(
    const float* __restrict__ q, const float* __restrict__ pts,
    int nq_total, int per_q, int npts, float r2,
    int* __restrict__ nbr, int* __restrict__ cnt)
{
  const int wave = blockIdx.x * (blockDim.x >> 5) + (threadIdx.x >> 5);
  if (wave >= nq_total) return;
  const int lane = threadIdx.x & 31;
  const int b = wave / per_q;
  const float* P = pts + (size_t)b * npts * 3;
  const float qx = q[(size_t)wave*3], qy = q[(size_t)wave*3+1], qz = q[(size_t)wave*3+2];
  int* nb = nbr + (size_t)wave * 64;
  int count = 0;
  for (int base = 0; base < npts; base += 32) {
    int j = base + lane;
    bool in = false;
    if (j < npts) {
      float dx = P[j*3] - qx, dy = P[j*3+1] - qy, dz = P[j*3+2] - qz;
      in = (dx*dx + dy*dy + dz*dz) <= r2;
    }
    unsigned mm = (unsigned)__ballot(in);                 // wave32 mask
    int pre = __popc(mm & ((1u << lane) - 1u));
    if (in && (count + pre) < 64) nb[count + pre] = j;
    count += __popc(mm);
  }
  if (lane == 0) cnt[wave] = count < 64 ? count : 64;
}

// ----- SA1: per center, relpos[64x3] -> MLP[3,64,64,128] -> masked max ------
__global__ void __launch_bounds__(128) sa1_kernel(
    const float* __restrict__ pos, const float* __restrict__ ctr1,
    const int* __restrict__ nbr, const int* __restrict__ cntArr,
    const _Float16* wh0, const float* b0, const _Float16* wh1, const float* b1,
    const _Float16* wh2, const float* b2, float* __restrict__ x1)
{
  __shared__ __align__(16) _Float16 A0[64*32];
  __shared__ __align__(16) _Float16 H1[64*64];
  __shared__ __align__(16) _Float16 H2[64*64];
  __shared__ __align__(16) _Float16 WT[16*64];
  __shared__ float OP[4*128];
  const int c = blockIdx.x;             // b*1024 + m
  const int b = c >> 10;
  const int tid = threadIdx.x;
  const int cnt = cntArr[c];
  const float cx = ctr1[(size_t)c*3], cy = ctr1[(size_t)c*3+1], cz = ctr1[(size_t)c*3+2];
  for (int i = tid; i < 64*32; i += blockDim.x) {
    int row = i >> 5, k = i & 31;
    float v = 0.0f;
    if (row < cnt && k < 3) {
      int j = nbr[(size_t)c*64 + row];
      v = pos[((size_t)b*NPTS + j)*3 + k] - (k == 0 ? cx : (k == 1 ? cy : cz));
    }
    A0[i] = (_Float16)v;
  }
  fused_layer(A0, 32, wh0, b0, 32,  64, H1, 64, nullptr, 0, WT, BN_SCALE);
  fused_layer(H1, 64, wh1, b1, 64,  64, H2, 64, nullptr, 0, WT, BN_SCALE);
  fused_layer(H2, 64, wh2, b2, 64, 128, nullptr, 0, OP, cnt, WT, BN_SCALE);
  for (int n = tid; n < 128; n += blockDim.x) {
    float v = fmaxf(fmaxf(OP[n], OP[128+n]), fmaxf(OP[256+n], OP[384+n]));
    x1[(size_t)c*128 + n] = v;
  }
}

// ----- SA2: per center, [x1|relpos][64x131] -> MLP[131,128,128,256] ---------
__global__ void __launch_bounds__(128) sa2_kernel(
    const float* __restrict__ x1, const float* __restrict__ ctr1,
    const float* __restrict__ ctr2,
    const int* __restrict__ nbr, const int* __restrict__ cntArr,
    const _Float16* wh0, const float* b0, const _Float16* wh1, const float* b1,
    const _Float16* wh2, const float* b2, float* __restrict__ x2)
{
  __shared__ __align__(16) _Float16 A0[64*160];
  __shared__ __align__(16) _Float16 H1[64*128];
  __shared__ __align__(16) _Float16 H2[64*128];
  __shared__ __align__(16) _Float16 WT[16*160];
  __shared__ float OP[4*256];
  const int c = blockIdx.x;             // b*256 + m
  const int b = c >> 8;
  const int tid = threadIdx.x;
  const int cnt = cntArr[c];
  for (int i = tid; i < 64*160; i += blockDim.x) {
    int row = i / 160, k = i - row*160;
    float v = 0.0f;
    if (row < cnt) {
      int j = nbr[(size_t)c*64 + row];    // index into stage-1 centers
      if (k < 128)      v = x1[((size_t)b*M1 + j)*128 + k];
      else if (k < 131) v = ctr1[((size_t)b*M1 + j)*3 + (k-128)]
                            - ctr2[(size_t)c*3 + (k-128)];
    }
    A0[i] = (_Float16)v;
  }
  fused_layer(A0, 160, wh0, b0, 160, 128, H1, 128, nullptr, 0, WT, BN_SCALE);
  fused_layer(H1, 128, wh1, b1, 128, 128, H2, 128, nullptr, 0, WT, BN_SCALE);
  fused_layer(H2, 128, wh2, b2, 128, 256, nullptr, 0, OP, cnt, WT, BN_SCALE);
  for (int n = tid; n < 256; n += blockDim.x) {
    float v = fmaxf(fmaxf(OP[n], OP[256+n]), fmaxf(OP[512+n], OP[768+n]));
    x2[(size_t)c*256 + n] = v;
  }
}

// ----- zero init for global max-pool buffer ---------------------------------
__global__ void zero_kernel(float* __restrict__ p, int n) {
  int i = blockIdx.x * blockDim.x + threadIdx.x;
  if (i < n) p[i] = 0.0f;
}

// ----- SA3: pointwise MLP[259,256,512,1024] + global max pool ---------------
// 32 points per block (2 M-tiles / 2 waves); LDS arena with aliasing <= 64KB.
__global__ void __launch_bounds__(64) sa3_kernel(
    const float* __restrict__ x2, const float* __restrict__ ctr2,
    const _Float16* wh0, const float* b0, const _Float16* wh1, const float* b1,
    const _Float16* wh2, const float* b2, float* __restrict__ g)
{
  __shared__ __align__(16) char arena[65536];
  _Float16* H2 = (_Float16*)arena;              // [32][512]  32768 B
  _Float16* H1 = (_Float16*)(arena + 32768);    // [32][256]  16384 B
  _Float16* WT = (_Float16*)(arena + 49152);    // [16][512]  16384 B
  _Float16* A0 = (_Float16*)arena;              // [32][288]  aliases H2 (dead)
  float*    OP = (float*)(arena + 32768);       // [2][1024]  aliases H1 (dead)
  const int blk = blockIdx.x;                   // 0..127
  const int b   = blk >> 3;
  const int p0  = (blk & 7) << 5;               // 32-point chunk
  const int tid = threadIdx.x;
  for (int i = tid; i < 32*288; i += blockDim.x) {
    int row = i / 288, k = i - row*288;
    int p = p0 + row;
    float v = 0.0f;
    if (k < 256)      v = x2[((size_t)b*M2 + p)*256 + k];
    else if (k < 259) v = ctr2[((size_t)b*M2 + p)*3 + (k-256)];
    A0[i] = (_Float16)v;
  }
  fused_layer(A0, 288, wh0, b0, 288,  256, H1, 256, nullptr, 0, WT, BN_SCALE);
  fused_layer(H1, 256, wh1, b1, 256,  512, H2, 512, nullptr, 0, WT, BN_SCALE);
  fused_layer(H2, 512, wh2, b2, 512, 1024, nullptr, 0, OP, 32, WT, BN_SCALE);
  for (int n = tid; n < 1024; n += blockDim.x) {
    float v = fmaxf(OP[n], OP[1024 + n]);       // values >= 0
    atomicMax((unsigned int*)&g[(size_t)b*1024 + n], __float_as_uint(v));
  }
}

// ----- classification head: one block per cloud -----------------------------
__global__ void __launch_bounds__(256) head_kernel(
    const float* __restrict__ g,
    const float* w1, const float* b1, const float* w2, const float* b2,
    const float* w3, const float* b3, float* __restrict__ out)
{
  const int b = blockIdx.x;
  const int tid = threadIdx.x;
  __shared__ float gin[1024];
  __shared__ float t1[512];
  __shared__ float t2[256];
  __shared__ float lg[40];
  for (int i = tid; i < 1024; i += 256) gin[i] = g[(size_t)b*1024 + i];
  __syncthreads();
  for (int o = tid; o < 512; o += 256) {
    float s = b1[o];
    for (int k = 0; k < 1024; ++k) s += gin[k] * w1[(size_t)k*512 + o];
    t1[o] = fmaxf(s, 0.0f);
  }
  __syncthreads();
  if (tid < 256) {
    float s = b2[tid];
    for (int k = 0; k < 512; ++k) s += t1[k] * w2[(size_t)k*256 + tid];
    t2[tid] = fmaxf(s, 0.0f);
  }
  __syncthreads();
  if (tid < 40) {
    float s = b3[tid];
    for (int k = 0; k < 256; ++k) s += t2[k] * w3[(size_t)k*40 + tid];
    lg[tid] = s;
  }
  __syncthreads();
  if (tid == 0) {
    float mx = -3.0e38f;
    for (int i = 0; i < 40; ++i) mx = fmaxf(mx, lg[i]);
    float se = 0.0f;
    for (int i = 0; i < 40; ++i) se += expf(lg[i] - mx);
    float lse = mx + logf(se);
    for (int i = 0; i < 40; ++i) out[(size_t)b*40 + i] = lg[i] - lse;
  }
}

// ---------------------------------------------------------------------------
extern "C" void kernel_launch(void* const* d_in, const int* in_sizes, int n_in,
                              void* d_out, int out_size, void* d_ws, size_t ws_size,
                              hipStream_t stream)
{
  (void)in_sizes; (void)n_in; (void)out_size; (void)ws_size;
  const float* pos  = (const float*)d_in[0];
  const float* s1w0 = (const float*)d_in[1];  const float* s1b0 = (const float*)d_in[2];
  const float* s1w1 = (const float*)d_in[3];  const float* s1b1 = (const float*)d_in[4];
  const float* s1w2 = (const float*)d_in[5];  const float* s1b2 = (const float*)d_in[6];
  const float* s2w0 = (const float*)d_in[7];  const float* s2b0 = (const float*)d_in[8];
  const float* s2w1 = (const float*)d_in[9];  const float* s2b1 = (const float*)d_in[10];
  const float* s2w2 = (const float*)d_in[11]; const float* s2b2 = (const float*)d_in[12];
  const float* s3w0 = (const float*)d_in[13]; const float* s3b0 = (const float*)d_in[14];
  const float* s3w1 = (const float*)d_in[15]; const float* s3b1 = (const float*)d_in[16];
  const float* s3w2 = (const float*)d_in[17]; const float* s3b2 = (const float*)d_in[18];
  const float* l1w  = (const float*)d_in[19]; const float* l1b  = (const float*)d_in[20];
  const float* l2w  = (const float*)d_in[21]; const float* l2b  = (const float*)d_in[22];
  const float* l3w  = (const float*)d_in[23]; const float* l3b  = (const float*)d_in[24];

  char* ws = (char*)d_ws;
  float* ctr1 = (float*)(ws + 0);          // 16*1024*3 f32
  float* ctr2 = (float*)(ws + 196608);     // 16*256*3  f32
  int*   nbr1 = (int*)  (ws + 245760);     // 16*1024*64 i32
  int*   cnt1 = (int*)  (ws + 4440064);    // 16*1024 i32
  int*   nbr2 = (int*)  (ws + 4505600);    // 16*256*64 i32
  int*   cnt2 = (int*)  (ws + 5554176);    // 16*256 i32
  float* x1   = (float*)(ws + 5570560);    // 16*1024*128 f32
  float* x2   = (float*)(ws + 13959168);   // 16*256*256 f32
  float* g    = (float*)(ws + 18153472);   // 16*1024 f32
  // Pre-converted f16 weights, [Nout][Kpad] N-major, zero-padded:
  _Float16* wh10 = (_Float16*)(ws + 18219008);   //  64x32
  _Float16* wh11 = (_Float16*)(ws + 18223104);   //  64x64
  _Float16* wh12 = (_Float16*)(ws + 18231296);   // 128x64
  _Float16* wh20 = (_Float16*)(ws + 18247680);   // 128x160
  _Float16* wh21 = (_Float16*)(ws + 18288640);   // 128x128
  _Float16* wh22 = (_Float16*)(ws + 18321408);   // 256x128
  _Float16* wh30 = (_Float16*)(ws + 18386944);   // 256x288
  _Float16* wh31 = (_Float16*)(ws + 18534400);   // 512x256
  _Float16* wh32 = (_Float16*)(ws + 18796544);   // 1024x512 (ends 19845120)

  auto cvt = [&](const float* W, int Kin, int Nout, int Kpad, _Float16* out) {
    int total = Nout * Kpad;
    convertw_kernel<<<(total + 255) / 256, 256, 0, stream>>>(W, Kin, Nout, Kpad, out);
  };
  cvt(s1w0,   3,   64,  32, wh10);
  cvt(s1w1,  64,   64,  64, wh11);
  cvt(s1w2,  64,  128,  64, wh12);
  cvt(s2w0, 131,  128, 160, wh20);
  cvt(s2w1, 128,  128, 128, wh21);
  cvt(s2w2, 128,  256, 128, wh22);
  cvt(s3w0, 259,  256, 288, wh30);
  cvt(s3w1, 256,  512, 256, wh31);
  cvt(s3w2, 512, 1024, 512, wh32);

  fps_kernel<<<BATCH, 256, 0, stream>>>(pos, NPTS, M1, ctr1);
  radius_kernel<<<(BATCH*M1)/4, 128, 0, stream>>>(ctr1, pos, BATCH*M1, M1, NPTS,
                                                  0.04f, nbr1, cnt1);
  sa1_kernel<<<BATCH*M1, 128, 0, stream>>>(pos, ctr1, nbr1, cnt1,
                                           wh10, s1b0, wh11, s1b1, wh12, s1b2, x1);
  fps_kernel<<<BATCH, 256, 0, stream>>>(ctr1, M1, M2, ctr2);
  radius_kernel<<<(BATCH*M2)/4, 128, 0, stream>>>(ctr2, ctr1, BATCH*M2, M2, M1,
                                                  0.16f, nbr2, cnt2);
  sa2_kernel<<<BATCH*M2, 128, 0, stream>>>(x1, ctr1, ctr2, nbr2, cnt2,
                                           wh20, s2b0, wh21, s2b1, wh22, s2b2, x2);
  zero_kernel<<<(BATCH*1024)/256, 256, 0, stream>>>(g, BATCH*1024);
  sa3_kernel<<<BATCH*8, 64, 0, stream>>>(x2, ctr2,
                                         wh30, s3b0, wh31, s3b1, wh32, s3b2, g);
  head_kernel<<<BATCH, 256, 0, stream>>>(g, l1w, l1b, l2w, l2b, l3w, l3b,
                                         (float*)d_out);
}
